// AffectiveMemoryUnit1D_39152921870436
// MI455X (gfx1250) — matrix operand order repaired and checked by reading, not compiled, not run on hardware
//
#include <hip/hip_runtime.h>

#define BB 8
#define NN 4096
#define DD 1024
#define KK 128
#define MTOT (BB * NN)
#define NCHUNK 32
#define CH (NN / NCHUNK)

// GEMM tiling: K-chunk staged through LDS (A and B), double-buffered async DMA
#define KC 64
#define LDSROW (KC + 8)        // 72 elems = 144B row stride -> conflict-free banks

typedef __attribute__((ext_vector_type(16))) __bf16 v16bf;
typedef __attribute__((ext_vector_type(8)))  float  v8f;

// vector-of-4-int type matching the async builtin's pointee per hipcc diagnostic
typedef int v4i_vs __attribute__((vector_size(4 * sizeof(int))));
typedef __attribute__((address_space(1))) v4i_vs gv4i;
typedef __attribute__((address_space(3))) v4i_vs lv4i;

union Frag { v16bf v; uint4 q[2]; };

#if defined(__has_builtin)
#  if __has_builtin(__builtin_amdgcn_global_load_async_to_lds_b128)
#    define HAVE_ASYNC 1
#  endif
#endif
#ifndef HAVE_ASYNC
#  define HAVE_ASYNC 0
#endif

__device__ __forceinline__ unsigned short f2bf(float f) {
  unsigned int u = __float_as_uint(f);
  unsigned int r = u + 0x7FFFu + ((u >> 16) & 1u);   // round-to-nearest-even
  return (unsigned short)(r >> 16);
}

// Wait until ASYNCcnt <= N (no-op in the non-async fallback: __syncthreads covers dscnt)
template <int N>
__device__ __forceinline__ void wait_async() {
#if HAVE_ASYNC
#  if __has_builtin(__builtin_amdgcn_s_wait_asynccnt)
  __builtin_amdgcn_s_wait_asynccnt(N);
#  else
  if (N == 0) asm volatile("s_wait_asynccnt 0x0" ::: "memory");
  else        asm volatile("s_wait_asynccnt 0x6" ::: "memory");
#  endif
#endif
}

// Stage a ROWS x KC bf16 tile (row stride Kd in global) into padded LDS.
// 256 threads; ROWS*8 b128 segments; global-contiguous within rows (coalesced).
template <int ROWS>
__device__ __forceinline__ void stage_tile(const unsigned short* __restrict__ gsrc,
                                           unsigned short* ldst, int tid, int Kd) {
#pragma unroll
  for (int i = 0; i < ROWS / 32; ++i) {
    const int s   = tid + i * 256;       // 0 .. ROWS*8-1
    const int row = s >> 3;              // 0 .. ROWS-1
    const int cs  = s & 7;               // 16-byte segment in row
    const unsigned short* g = gsrc + (size_t)row * Kd + cs * 8;
    unsigned short*       l = ldst + row * LDSROW + cs * 8;
#if HAVE_ASYNC
    __builtin_amdgcn_global_load_async_to_lds_b128((gv4i*)g, (lv4i*)l, 0, 0);
#else
    *(uint4*)l = *(const uint4*)g;
#endif
  }
}

// ---------------------------------------------------------------------------
// fp32 -> bf16 conversion (weights)
// ---------------------------------------------------------------------------
__global__ __launch_bounds__(256)
void f32_to_bf16_kernel(const float* __restrict__ in, unsigned short* __restrict__ out, int n) {
  int i = blockIdx.x * 256 + threadIdx.x;
  if (i < n) out[i] = f2bf(in[i]);
}

// ---------------------------------------------------------------------------
// LayerNorm over D=1024, output bf16. One 256-thread block per row.
// ---------------------------------------------------------------------------
__global__ __launch_bounds__(256)
void ln_to_bf16_kernel(const float* __restrict__ x, const float* __restrict__ g,
                       const float* __restrict__ b, unsigned short* __restrict__ out) {
  const int row = blockIdx.x;
  const int t   = threadIdx.x;
  const float4 v = ((const float4*)(x + (size_t)row * DD))[t];
  __shared__ float r1[256], r2[256];
  r1[t] = v.x + v.y + v.z + v.w;
  r2[t] = v.x * v.x + v.y * v.y + v.z * v.z + v.w * v.w;
  __syncthreads();
  for (int o = 128; o > 0; o >>= 1) {
    if (t < o) { r1[t] += r1[t + o]; r2[t] += r2[t + o]; }
    __syncthreads();
  }
  const float mean = r1[0] * (1.0f / DD);
  const float var  = r2[0] * (1.0f / DD) - mean * mean;
  const float inv  = rsqrtf(var + 1e-5f);
  const float4 gv = ((const float4*)g)[t];
  const float4 bv = ((const float4*)b)[t];
  ushort4 o4;
  o4.x = f2bf((v.x - mean) * inv * gv.x + bv.x);
  o4.y = f2bf((v.y - mean) * inv * gv.y + bv.y);
  o4.z = f2bf((v.z - mean) * inv * gv.z + bv.z);
  o4.w = f2bf((v.w - mean) * inv * gv.w + bv.w);
  ((ushort4*)(out + (size_t)row * DD))[t] = o4;
}

// ---------------------------------------------------------------------------
// WMMA bf16 GEMM:  C[m][n] = sum_k A[m][k] * Bw[n][k]  (+ bias[n])
// Block = 8 waves; block tile 128M x 64N; per-wave 16M x 64N.
// Both A and B staged global->LDS via async DMA, double-buffered per KC chunk.
// Steady-state inner loop: ds_load_b128 + v_wmma only (no global loads).
// ---------------------------------------------------------------------------
template <bool OUT_BF16, bool HAS_BIAS>
__global__ __launch_bounds__(256)
void gemm_wmma_kernel(const unsigned short* __restrict__ A,
                      const unsigned short* __restrict__ Bw,
                      const float* __restrict__ bias,
                      void* __restrict__ Cout, int Ncols, int Kd) {
  __shared__ unsigned short la[2][128 * LDSROW];   // 2 x 18KB
  __shared__ unsigned short lb[2][64 * LDSROW];    // 2 x  9KB
  const int tid  = threadIdx.x;
  const int lane = tid & 31;
  const int wave = tid >> 5;
  const int r    = lane & 15;
  const int hi   = lane >> 4;
  const int n0   = blockIdx.x * 64;
  const int NCk  = Kd / KC;

  const unsigned short* abase = A  + (size_t)(blockIdx.y * 128) * Kd;
  const unsigned short* bbase = Bw + (size_t)n0 * Kd;

  // chunk 0 in flight
  stage_tile<128>(abase, &la[0][0], tid, Kd);
  stage_tile<64>(bbase, &lb[0][0], tid, Kd);

  v8f acc0 = {}, acc1 = {}, acc2 = {}, acc3 = {};

  for (int c = 0; c < NCk; ++c) {
    const int kb = c * KC;
    const bool more = (c + 1 < NCk);
    if (more) {
      stage_tile<128>(abase + kb + KC, &la[(c + 1) & 1][0], tid, Kd);   // 4 issues
      stage_tile<64>(bbase + kb + KC, &lb[(c + 1) & 1][0], tid, Kd);    // 2 issues
      wait_async<6>();          // retire current chunk; next chunk stays in flight
    } else {
      wait_async<0>();
    }
    __syncthreads();            // chunk c visible; buffer (c+1)&1 free for DMA writers
    if (more && c + 2 < NCk)    // warm L2 for chunk c+2 (global_prefetch_b8)
      __builtin_prefetch(abase + (size_t)(tid & 127) * Kd + kb + 2 * KC, 0, 1);

    // A fragment (16x32): lane<16 holds K 0..15 contiguous, lane>=16 holds 16..31
    const unsigned short* apl = &la[c & 1][(wave * 16 + r) * LDSROW + hi * 16];
    // B fragment (32x16): lane<16 holds K 0..15 contiguous, lane>=16 holds 16..31
    const unsigned short* lkb = &lb[c & 1][hi * 16];

#pragma unroll
    for (int s = 0; s < KC / 32; ++s) {
      Frag fa;
      fa.q[0] = *(const uint4*)(apl + s * 32);
      fa.q[1] = *(const uint4*)(apl + s * 32 + 8);
      const unsigned short* lk = lkb + s * 32;
      Frag fb0, fb1, fb2, fb3;
      fb0.q[0] = *(const uint4*)(lk + (0 * 16 + r) * LDSROW);
      fb0.q[1] = *(const uint4*)(lk + (0 * 16 + r) * LDSROW + 8);
      fb1.q[0] = *(const uint4*)(lk + (1 * 16 + r) * LDSROW);
      fb1.q[1] = *(const uint4*)(lk + (1 * 16 + r) * LDSROW + 8);
      fb2.q[0] = *(const uint4*)(lk + (2 * 16 + r) * LDSROW);
      fb2.q[1] = *(const uint4*)(lk + (2 * 16 + r) * LDSROW + 8);
      fb3.q[0] = *(const uint4*)(lk + (3 * 16 + r) * LDSROW);
      fb3.q[1] = *(const uint4*)(lk + (3 * 16 + r) * LDSROW + 8);
      acc0 = __builtin_amdgcn_wmma_f32_16x16x32_bf16(false, fa.v, false, fb0.v, (short)0, acc0, false, false);
      acc1 = __builtin_amdgcn_wmma_f32_16x16x32_bf16(false, fa.v, false, fb1.v, (short)0, acc1, false, false);
      acc2 = __builtin_amdgcn_wmma_f32_16x16x32_bf16(false, fa.v, false, fb2.v, (short)0, acc2, false, false);
      acc3 = __builtin_amdgcn_wmma_f32_16x16x32_bf16(false, fa.v, false, fb3.v, (short)0, acc3, false, false);
    }
    __syncthreads();            // done reading buffer c&1 before DMA overwrites it
  }

  const int nb = n0 + r;
  float b0 = 0.f, b1 = 0.f, b2 = 0.f, b3 = 0.f;
  if (HAS_BIAS) { b0 = bias[nb]; b1 = bias[nb + 16]; b2 = bias[nb + 32]; b3 = bias[nb + 48]; }
  const int mrow = blockIdx.y * 128 + wave * 16 + hi * 8;
  if (OUT_BF16) {
    unsigned short* C = (unsigned short*)Cout;
#pragma unroll
    for (int e = 0; e < 8; ++e) {
      size_t o = (size_t)(mrow + e) * Ncols + nb;
      C[o]      = f2bf(acc0[e] + b0);
      C[o + 16] = f2bf(acc1[e] + b1);
      C[o + 32] = f2bf(acc2[e] + b2);
      C[o + 48] = f2bf(acc3[e] + b3);
    }
  } else {
    float* C = (float*)Cout;
#pragma unroll
    for (int e = 0; e < 8; ++e) {
      size_t o = (size_t)(mrow + e) * Ncols + nb;
      C[o]      = acc0[e] + b0;
      C[o + 16] = acc1[e] + b1;
      C[o + 32] = acc2[e] + b2;
      C[o + 48] = acc3[e] + b3;
    }
  }
}

// ---------------------------------------------------------------------------
// Softmax over the N axis (per (b,k) column): chunked online softmax.
// ---------------------------------------------------------------------------
__global__ __launch_bounds__(KK)
void softmax_part_kernel(const float* __restrict__ a, float* __restrict__ pm,
                         float* __restrict__ ps) {
  const int b = blockIdx.y, chunk = blockIdx.x, k = threadIdx.x;
  const float* p = a + ((size_t)b * NN + (size_t)chunk * CH) * KK + k;
  float m = p[0];
  float s = 1.0f;
  for (int i = 1; i < CH; ++i) {
    float xv = p[(size_t)i * KK];
    float mn = fmaxf(m, xv);
    s = s * __expf(m - mn) + __expf(xv - mn);
    m = mn;
  }
  const int o = (b * NCHUNK + chunk) * KK + k;
  pm[o] = m;
  ps[o] = s;
}

__global__ __launch_bounds__(KK)
void softmax_combine_kernel(const float* __restrict__ pm, const float* __restrict__ ps,
                            float* __restrict__ gm, float* __restrict__ gs) {
  const int b = blockIdx.x, k = threadIdx.x;
  float m = pm[(b * NCHUNK) * KK + k];
  float s = ps[(b * NCHUNK) * KK + k];
  for (int c = 1; c < NCHUNK; ++c) {
    float mm = pm[(b * NCHUNK + c) * KK + k];
    float ss = ps[(b * NCHUNK + c) * KK + k];
    float mn = fmaxf(m, mm);
    s = s * __expf(m - mn) + ss * __expf(mm - mn);
    m = mn;
  }
  gm[b * KK + k] = m;
  gs[b * KK + k] = s;
}

// softmax value + L1 normalize over K, emit bf16 attention weights
__global__ __launch_bounds__(KK)
void softmax_norm_kernel(const float* __restrict__ a, const float* __restrict__ gm,
                         const float* __restrict__ gs, unsigned short* __restrict__ aout) {
  const int row = blockIdx.x;           // 0..MTOT-1 (= b*NN + n)
  const int b   = row / NN;
  const int k   = threadIdx.x;
  float p = __expf(a[(size_t)row * KK + k] - gm[b * KK + k]) / gs[b * KK + k];
  __shared__ float sr[KK];
  sr[k] = p;
  __syncthreads();
  for (int o = KK / 2; o > 0; o >>= 1) {
    if (k < o) sr[k] += sr[k + o];
    __syncthreads();
  }
  p = p / (1e-9f + sr[0]);
  aout[(size_t)row * KK + k] = f2bf(p);
}

// ---------------------------------------------------------------------------
// Final: out = relu(LN(h4)*g + b + x)
// ---------------------------------------------------------------------------
__global__ __launch_bounds__(256)
void final_ln_res_kernel(const float* __restrict__ h, const float* __restrict__ xres,
                         const float* __restrict__ g, const float* __restrict__ b,
                         float* __restrict__ out) {
  const int row = blockIdx.x;
  const int t   = threadIdx.x;
  const float4 v = ((const float4*)(h + (size_t)row * DD))[t];
  __shared__ float r1[256], r2[256];
  r1[t] = v.x + v.y + v.z + v.w;
  r2[t] = v.x * v.x + v.y * v.y + v.z * v.z + v.w * v.w;
  __syncthreads();
  for (int o = 128; o > 0; o >>= 1) {
    if (t < o) { r1[t] += r1[t + o]; r2[t] += r2[t + o]; }
    __syncthreads();
  }
  const float mean = r1[0] * (1.0f / DD);
  const float var  = r2[0] * (1.0f / DD) - mean * mean;
  const float inv  = rsqrtf(var + 1e-5f);
  const float4 gv = ((const float4*)g)[t];
  const float4 bv = ((const float4*)b)[t];
  const float4 xv = ((const float4*)(xres + (size_t)row * DD))[t];
  float4 y;
  y.x = fmaxf((v.x - mean) * inv * gv.x + bv.x + xv.x, 0.f);
  y.y = fmaxf((v.y - mean) * inv * gv.y + bv.y + xv.y, 0.f);
  y.z = fmaxf((v.z - mean) * inv * gv.z + bv.z + xv.z, 0.f);
  y.w = fmaxf((v.w - mean) * inv * gv.w + bv.w + xv.w, 0.f);
  ((float4*)(out + (size_t)row * DD))[t] = y;
}

// ---------------------------------------------------------------------------
extern "C" void kernel_launch(void* const* d_in, const int* in_sizes, int n_in,
                              void* d_out, int out_size, void* d_ws, size_t ws_size,
                              hipStream_t stream) {
  (void)in_sizes; (void)n_in; (void)out_size; (void)ws_size;
  const float* x     = (const float*)d_in[0];
  const float* ln_g  = (const float*)d_in[1];
  const float* ln_b  = (const float*)d_in[2];
  const float* w_in  = (const float*)d_in[3];
  const float* b_in  = (const float*)d_in[4];
  const float* w0    = (const float*)d_in[5];
  const float* w1    = (const float*)d_in[6];
  const float* w_out = (const float*)d_in[7];
  const float* oln_g = (const float*)d_in[8];
  const float* oln_b = (const float*)d_in[9];
  float* out = (float*)d_out;

  char* ws = (char*)d_ws;
  size_t off = 0;
  auto take = [&](size_t bytes) -> char* {
    char* p = ws + off;
    off = (off + bytes + 255) & ~(size_t)255;
    return p;
  };

  unsigned short* w_in_bf  = (unsigned short*)take((size_t)DD * DD * 2);
  unsigned short* w0_bf    = (unsigned short*)take((size_t)KK * DD * 2);
  unsigned short* w1_bf    = (unsigned short*)take((size_t)DD * KK * 2);
  unsigned short* w_out_bf = (unsigned short*)take((size_t)DD * DD * 2);
  unsigned short* hln      = (unsigned short*)take((size_t)MTOT * DD * 2);  // reused as h3
  unsigned short* h2       = (unsigned short*)take((size_t)MTOT * DD * 2);
  float*          logits   = (float*)take((size_t)MTOT * KK * 4);
  unsigned short* aprime   = (unsigned short*)take((size_t)MTOT * KK * 2);
  float*          h4       = (float*)take((size_t)MTOT * DD * 4);
  float*          pm       = (float*)take((size_t)BB * NCHUNK * KK * 4);
  float*          ps       = (float*)take((size_t)BB * NCHUNK * KK * 4);
  float*          gm       = (float*)take((size_t)BB * KK * 4);
  float*          gs       = (float*)take((size_t)BB * KK * 4);

  // weights -> bf16 (tiny, L2 resident afterwards)
  f32_to_bf16_kernel<<<(DD * DD + 255) / 256, 256, 0, stream>>>(w_in, w_in_bf, DD * DD);
  f32_to_bf16_kernel<<<(KK * DD + 255) / 256, 256, 0, stream>>>(w0, w0_bf, KK * DD);
  f32_to_bf16_kernel<<<(DD * KK + 255) / 256, 256, 0, stream>>>(w1, w1_bf, DD * KK);
  f32_to_bf16_kernel<<<(DD * DD + 255) / 256, 256, 0, stream>>>(w_out, w_out_bf, DD * DD);

  // LN(x) -> bf16
  ln_to_bf16_kernel<<<MTOT, 256, 0, stream>>>(x, ln_g, ln_b, hln);

  // h2 = LN(x) @ w_in^T + b_in        (32768x1024 * 1024x1024)
  gemm_wmma_kernel<true, true><<<dim3(DD / 64, MTOT / 128), 256, 0, stream>>>(
      hln, w_in_bf, b_in, (void*)h2, DD, DD);
  // logits = h2 @ w0^T                (32768x1024 * 1024x128), fp32 out
  gemm_wmma_kernel<false, false><<<dim3(KK / 64, MTOT / 128), 256, 0, stream>>>(
      h2, w0_bf, nullptr, (void*)logits, KK, DD);

  // softmax over N per (b,k), then L1 over K per (b,n)
  softmax_part_kernel<<<dim3(NCHUNK, BB), KK, 0, stream>>>(logits, pm, ps);
  softmax_combine_kernel<<<BB, KK, 0, stream>>>(pm, ps, gm, gs);
  softmax_norm_kernel<<<MTOT, KK, 0, stream>>>(logits, gm, gs, aprime);

  // h3 = a' @ w1^T                    (32768x128 * 128x1024) -> reuse hln buffer
  gemm_wmma_kernel<true, false><<<dim3(DD / 64, MTOT / 128), 256, 0, stream>>>(
      aprime, w1_bf, nullptr, (void*)hln, DD, KK);
  // h4 = h3 @ w_out^T                 (32768x1024 * 1024x1024), fp32 out
  gemm_wmma_kernel<false, false><<<dim3(DD / 64, MTOT / 128), 256, 0, stream>>>(
      hln, w_out_bf, nullptr, (void*)h4, DD, DD);

  // out = relu(LN(h4) + x)
  final_ln_res_kernel<<<MTOT, 256, 0, stream>>>(h4, x, oln_g, oln_b, out);
}
